// QuantQwen3MLP_17025250361715
// MI455X (gfx1250) — 8-bit matrix operand order repaired and checked by reading, compiled, not run on hardware
//
#include <hip/hip_runtime.h>
#include <hip/hip_bf16.h>


#define TOK 8192
#define HD  2048
#define ID  6144

typedef int v8i __attribute__((ext_vector_type(8)));

union FragA { v8i v; int2 d[4]; };
union FragB { v8i v; int4 q[2]; };
union AccI  { v8i v; int  i[8]; };

__device__ inline int quant8(float x, float inv) {
    int q = (int)rintf(x * inv);
    q = q > 127 ? 127 : q;
    q = q < -127 ? -127 : q;
    return q;
}

__global__ __launch_bounds__(256)
void rowquant_kernel(const float* __restrict__ src,
                     signed char* __restrict__ qdst,
                     float* __restrict__ scales,
                     int cols) {
    const int row = blockIdx.x;
    const int tid = threadIdx.x;
    const int cols4 = cols >> 2;
    const float4* __restrict__ p = (const float4*)(src + (size_t)row * cols);

    __shared__ float red[256];
    float m = 0.0f;
    for (int i = tid; i < cols4; i += 256) {
        float4 v = p[i];
        m = fmaxf(m, fmaxf(fmaxf(fabsf(v.x), fabsf(v.y)),
                           fmaxf(fabsf(v.z), fabsf(v.w))));
    }
    red[tid] = m;
    __syncthreads();
    for (int s = 128; s > 0; s >>= 1) {
        if (tid < s) red[tid] = fmaxf(red[tid], red[tid + s]);
        __syncthreads();
    }
    const float scale = fmaxf(red[0] * (1.0f / 127.0f), 1e-8f);
    if (tid == 0) scales[row] = scale;
    const float inv = 1.0f / scale;

    unsigned int* __restrict__ qp =
        (unsigned int*)(qdst + (size_t)row * cols);
    for (int i = tid; i < cols4; i += 256) {
        float4 v = p[i];
        unsigned int b0 = (unsigned int)quant8(v.x, inv) & 255u;
        unsigned int b1 = (unsigned int)quant8(v.y, inv) & 255u;
        unsigned int b2 = (unsigned int)quant8(v.z, inv) & 255u;
        unsigned int b3 = (unsigned int)quant8(v.w, inv) & 255u;
        qp[i] = b0 | (b1 << 8) | (b2 << 16) | (b3 << 24);
    }
}

__device__ inline FragA load_fragA(const signed char* sA, int rowLocal, int lane) {
    FragA a;
    const int rl = rowLocal + (lane & 15);
    const int kb = (lane >> 4) * 16;
    const int2* pa = (const int2*)(sA + rl * 64 + kb);
    a.d[0] = pa[0]; a.d[1] = pa[1]; a.d[2] = pa[4]; a.d[3] = pa[5];
    return a;
}

__device__ inline FragB load_fragB(const signed char* sB, int colLocal, int lane) {
    FragB b;
    const int nr = colLocal + (lane & 15);
    const int kh = (lane >> 4) * 16;
    const int4* pb = (const int4*)(sB + nr * 64 + kh);
    b.q[0] = pb[0]; b.q[1] = pb[2];
    return b;
}

__global__ __launch_bounds__(256)
void gemm1_gateup_kernel(const signed char* __restrict__ qx,
                         const float* __restrict__ sx,
                         const signed char* __restrict__ qwg,
                         const float* __restrict__ swg,
                         const signed char* __restrict__ qwu,
                         const float* __restrict__ swu,
                         float* __restrict__ h) {
    __shared__ alignas(16) signed char sA[2][64 * 64];
    __shared__ alignas(16) signed char sBg[2][128 * 64];
    __shared__ alignas(16) signed char sBu[2][128 * 64];

    const int tid  = threadIdx.x;
    const int lane = tid & 31;
    const int wave = tid >> 5;
    const int wm   = wave & 3;
    const int wn   = wave >> 2;
    const int m0   = blockIdx.y * 64;
    const int n0   = blockIdx.x * 128;

    const int tr = tid >> 2;
    const int tk = (tid & 3) * 16;

    const signed char* gA  = qx  + (size_t)(m0 + tr)      * HD + tk;
    const signed char* gG0 = qwg + (size_t)(n0 + tr)      * HD + tk;
    const signed char* gG1 = qwg + (size_t)(n0 + tr + 64) * HD + tk;
    const signed char* gU0 = qwu + (size_t)(n0 + tr)      * HD + tk;
    const signed char* gU1 = qwu + (size_t)(n0 + tr + 64) * HD + tk;

    AccI accG[4], accU[4];
    for (int s = 0; s < 4; ++s)
        for (int j = 0; j < 8; ++j) { accG[s].i[j] = 0; accU[s].i[j] = 0; }

    *(int4*)(sA[0]  + tr * 64 + tk)        = *(const int4*)gA;
    *(int4*)(sBg[0] + tr * 64 + tk)        = *(const int4*)gG0;
    *(int4*)(sBg[0] + (tr + 64) * 64 + tk) = *(const int4*)gG1;
    *(int4*)(sBu[0] + tr * 64 + tk)        = *(const int4*)gU0;
    *(int4*)(sBu[0] + (tr + 64) * 64 + tk) = *(const int4*)gU1;
    __syncthreads();

    const int nIter = HD / 64;
    for (int it = 0; it < nIter; ++it) {
        const int cur = it & 1;
        const int nxt = cur ^ 1;
        const bool hasNext = (it + 1) < nIter;

        int4 rA, rG0, rG1, rU0, rU1;
        if (hasNext) {
            const int kk = (it + 1) * 64;
            rA  = *(const int4*)(gA  + kk);
            rG0 = *(const int4*)(gG0 + kk);
            rG1 = *(const int4*)(gG1 + kk);
            rU0 = *(const int4*)(gU0 + kk);
            rU1 = *(const int4*)(gU1 + kk);
            if (kk + 64 < HD) {
                __builtin_prefetch(gA  + kk + 64, 0, 3);
                __builtin_prefetch(gG0 + kk + 64, 0, 3);
                __builtin_prefetch(gU0 + kk + 64, 0, 3);
            }
        }

        FragA a = load_fragA(sA[cur], wm * 16, lane);
        FragB bg[4], bu[4];
        for (int s = 0; s < 4; ++s) {
            bg[s] = load_fragB(sBg[cur], wn * 64 + s * 16, lane);
            bu[s] = load_fragB(sBu[cur], wn * 64 + s * 16, lane);
        }
        for (int s = 0; s < 4; ++s) {
            accG[s].v = __builtin_amdgcn_wmma_i32_16x16x64_iu8(
                true, a.v, true, bg[s].v, accG[s].v, false, false);
            accU[s].v = __builtin_amdgcn_wmma_i32_16x16x64_iu8(
                true, a.v, true, bu[s].v, accU[s].v, false, false);
        }

        if (hasNext) {
            *(int4*)(sA[nxt]  + tr * 64 + tk)        = rA;
            *(int4*)(sBg[nxt] + tr * 64 + tk)        = rG0;
            *(int4*)(sBg[nxt] + (tr + 64) * 64 + tk) = rG1;
            *(int4*)(sBu[nxt] + tr * 64 + tk)        = rU0;
            *(int4*)(sBu[nxt] + (tr + 64) * 64 + tk) = rU1;
        }
        __syncthreads();
    }

    const int nlane = lane & 15;
    const int mhalf = (lane >> 4) * 8;
    for (int s = 0; s < 4; ++s) {
        const int n = n0 + wn * 64 + s * 16 + nlane;
        const float sg = swg[n];
        const float su = swu[n];
        for (int r = 0; r < 8; ++r) {
            const int m = m0 + wm * 16 + mhalf + r;
            const float a0 = sx[m];
            const float g  = a0 * sg * (float)accG[s].i[r];
            const float u  = a0 * su * (float)accU[s].i[r];
            const float gs = g * (1.0f / (1.0f + __expf(-g)));
            h[(size_t)m * ID + n] = gs * u;
        }
    }
}

__global__ __launch_bounds__(256)
void gemm2_down_kernel(const signed char* __restrict__ qh,
                       const float* __restrict__ sh,
                       const signed char* __restrict__ qwd,
                       const float* __restrict__ swd,
                       float* __restrict__ out) {
    __shared__ alignas(16) signed char sA[2][64 * 64];
    __shared__ alignas(16) signed char sB[2][128 * 64];

    const int tid  = threadIdx.x;
    const int lane = tid & 31;
    const int wave = tid >> 5;
    const int wm   = wave & 3;
    const int wn   = wave >> 2;
    const int m0   = blockIdx.y * 64;
    const int n0   = blockIdx.x * 128;

    const int tr = tid >> 2;
    const int tk = (tid & 3) * 16;

    const signed char* gA  = qh  + (size_t)(m0 + tr)      * ID + tk;
    const signed char* gB0 = qwd + (size_t)(n0 + tr)      * ID + tk;
    const signed char* gB1 = qwd + (size_t)(n0 + tr + 64) * ID + tk;

    AccI acc[4];
    for (int s = 0; s < 4; ++s)
        for (int j = 0; j < 8; ++j) acc[s].i[j] = 0;

    *(int4*)(sA[0] + tr * 64 + tk)        = *(const int4*)gA;
    *(int4*)(sB[0] + tr * 64 + tk)        = *(const int4*)gB0;
    *(int4*)(sB[0] + (tr + 64) * 64 + tk) = *(const int4*)gB1;
    __syncthreads();

    const int nIter = ID / 64;
    for (int it = 0; it < nIter; ++it) {
        const int cur = it & 1;
        const int nxt = cur ^ 1;
        const bool hasNext = (it + 1) < nIter;

        int4 rA, rB0, rB1;
        if (hasNext) {
            const int kk = (it + 1) * 64;
            rA  = *(const int4*)(gA  + kk);
            rB0 = *(const int4*)(gB0 + kk);
            rB1 = *(const int4*)(gB1 + kk);
            if (kk + 64 < ID) {
                __builtin_prefetch(gA  + kk + 64, 0, 3);
                __builtin_prefetch(gB0 + kk + 64, 0, 3);
            }
        }

        FragA a = load_fragA(sA[cur], wm * 16, lane);
        FragB b[4];
        for (int s = 0; s < 4; ++s)
            b[s] = load_fragB(sB[cur], wn * 64 + s * 16, lane);
        for (int s = 0; s < 4; ++s)
            acc[s].v = __builtin_amdgcn_wmma_i32_16x16x64_iu8(
                true, a.v, true, b[s].v, acc[s].v, false, false);

        if (hasNext) {
            *(int4*)(sA[nxt] + tr * 64 + tk)        = rA;
            *(int4*)(sB[nxt] + tr * 64 + tk)        = rB0;
            *(int4*)(sB[nxt] + (tr + 64) * 64 + tk) = rB1;
        }
        __syncthreads();
    }

    const int nlane = lane & 15;
    const int mhalf = (lane >> 4) * 8;
    for (int s = 0; s < 4; ++s) {
        const int n = n0 + wn * 64 + s * 16 + nlane;
        const float sd = swd[n];
        for (int r = 0; r < 8; ++r) {
            const int m = m0 + wm * 16 + mhalf + r;
            out[(size_t)m * HD + n] = sh[m] * sd * (float)acc[s].i[r];
        }
    }
}

extern "C" void kernel_launch(void* const* d_in, const int* in_sizes, int n_in,
                              void* d_out, int out_size, void* d_ws, size_t ws_size,
                              hipStream_t stream) {
    const float* x      = (const float*)d_in[0];
    const float* w_gate = (const float*)d_in[1];
    const float* w_up   = (const float*)d_in[2];
    const float* w_down = (const float*)d_in[3];
    float* outp = (float*)d_out;

    char* ws = (char*)d_ws;
    size_t off = 0;
    auto take = [&](size_t bytes) -> char* {
        char* p = ws + off;
        off += (bytes + 255) & ~(size_t)255;
        return p;
    };
    signed char* q_x  = (signed char*)take((size_t)TOK * HD);
    float*       s_x  = (float*)take((size_t)TOK * 4);
    signed char* q_wg = (signed char*)take((size_t)ID * HD);
    float*       s_wg = (float*)take((size_t)ID * 4);
    signed char* q_wu = (signed char*)take((size_t)ID * HD);
    float*       s_wu = (float*)take((size_t)ID * 4);
    signed char* q_wd = (signed char*)take((size_t)HD * ID);
    float*       s_wd = (float*)take((size_t)HD * 4);
    float*       hbuf = (float*)take((size_t)TOK * ID * 4);
    signed char* q_h  = (signed char*)take((size_t)TOK * ID);
    float*       s_h  = (float*)take((size_t)TOK * 4);

    const dim3 blk(256);

    rowquant_kernel<<<TOK, blk, 0, stream>>>(x,      q_x,  s_x,  HD);
    rowquant_kernel<<<ID,  blk, 0, stream>>>(w_gate, q_wg, s_wg, HD);
    rowquant_kernel<<<ID,  blk, 0, stream>>>(w_up,   q_wu, s_wu, HD);
    rowquant_kernel<<<HD,  blk, 0, stream>>>(w_down, q_wd, s_wd, ID);

    gemm1_gateup_kernel<<<dim3(ID / 128, TOK / 64), blk, 0, stream>>>(
        q_x, s_x, q_wg, s_wg, q_wu, s_wu, hbuf);

    rowquant_kernel<<<TOK, blk, 0, stream>>>(hbuf, q_h, s_h, ID);

    gemm2_down_kernel<<<dim3(HD / 128, TOK / 64), blk, 0, stream>>>(
        q_h, s_h, q_wd, s_wd, outp);
}
